// SE3Unet_66864050864664
// MI455X (gfx1250) — matrix operand order repaired and checked by reading, not compiled, not run on hardware
//
#include <hip/hip_runtime.h>
#include <hip/hip_bf16.h>
#include <math.h>

typedef __attribute__((ext_vector_type(16))) __bf16 v16bf;
typedef __attribute__((ext_vector_type(8)))  float  v8f;
typedef __attribute__((ext_vector_type(4)))  unsigned int u32x4;
typedef __attribute__((ext_vector_type(8)))  int i32x8;
typedef __attribute__((ext_vector_type(4)))  int i32x4;

#define N0c 32768
#define N1c 8192
#define N2c 2048
#define DEGc 8
#define E0c (N0c * DEGc)
#define E1c (N1c * DEGc)
#define E2c (N2c * DEGc)
#define CCH 32
#define OUTc 40
#define EPSc 1e-9f
#define TPW 4   // row tiles per wave in the WMMA GEMM

static inline int cdiv(long long a, int b) { return (int)((a + b - 1) / b); }

// Feature layout (ALL feature buffers): F[(node*3 + vcomp)*Cc + c]
// => per-row contiguous 32 channels; per-node contiguous 96 floats.

// ---------------------------------------------------------------------------
// Geometry
// ---------------------------------------------------------------------------
__global__ void k_edge_geo(const float* __restrict__ pos,
                           const int* __restrict__ src, const int* __restrict__ dst,
                           float* __restrict__ rel, float* __restrict__ r, int E) {
  int e = blockIdx.x * blockDim.x + threadIdx.x;
  if (e >= E) return;
  int s = src[e], d = dst[e];
  float dx = pos[d * 3 + 0] - pos[s * 3 + 0];
  float dy = pos[d * 3 + 1] - pos[s * 3 + 1];
  float dz = pos[d * 3 + 2] - pos[s * 3 + 2];
  rel[e * 3 + 0] = dx; rel[e * 3 + 1] = dy; rel[e * 3 + 2] = dz;
  r[e] = sqrtf(dx * dx + dy * dy + dz * dz + 1e-12f);
}

__global__ void k_gather_pos(const float* __restrict__ pin, const int* __restrict__ idx,
                             float* __restrict__ pout, int n) {
  int i = blockIdx.x * blockDim.x + threadIdx.x;
  if (i >= n) return;
  int j = idx[i];
  pout[i * 3 + 0] = pin[j * 3 + 0];
  pout[i * 3 + 1] = pin[j * 3 + 1];
  pout[i * 3 + 2] = pin[j * 3 + 2];
}

// ---------------------------------------------------------------------------
// WMMA channel-mixing linear: X[n*3,32] @ W[32,32] for 4 weights (q,k,v,skip)
// Weights staged to LDS by the Tensor Data Mover; B fragments persist in
// VGPRs across a grid-stride loop over 16-row tiles (rows % 16 == 0 always).
// ---------------------------------------------------------------------------
__global__ __launch_bounds__(256)
void k_linear4_wmma(const float* __restrict__ x,
                    const float* __restrict__ Wq, const float* __restrict__ Wk,
                    const float* __restrict__ Wv, const float* __restrict__ Ws,
                    float* __restrict__ oq, float* __restrict__ ok,
                    float* __restrict__ ov, float* __restrict__ osk, int n) {
  __shared__ float Wlf[4][CCH][CCH];   // [which][k_in][k_out], f32
  int tid = threadIdx.x;

#if __has_builtin(__builtin_amdgcn_tensor_load_to_lds)
  if (tid < 32) {   // wave 0 drives the TDM; TENSORcnt is per-wave
    unsigned lbase = (unsigned)(uintptr_t)(&Wlf[0][0][0]);  // low 32b of generic = LDS offset
    const float* Wsrc[4] = {Wq, Wk, Wv, Ws};
#pragma unroll
    for (int w = 0; w < 4; ++w) {
      unsigned long long ga = (unsigned long long)(uintptr_t)Wsrc[w];
      u32x4 g0;
      g0[0] = 1u;                                   // count=1, user mode
      g0[1] = lbase + (unsigned)w * 4096u;          // lds_addr (bytes)
      g0[2] = (unsigned)ga;                         // global_addr[31:0]
      g0[3] = (unsigned)((ga >> 32) & 0x01FFFFFFull) | (2u << 30); // addr[56:32] | type=2
      i32x8 g1;
      g1[0] = 0x00020000;                           // data_size=2 (4B), no multicast
      g1[1] = (int)(1024u << 16);                   // tensor_dim0 = 1024 (low16)
      g1[2] = 0;                                    // dim0 hi16 / dim1 lo16
      g1[3] = (int)(1024u << 16);                   // tile_dim0 = 1024 (1-D tile)
      g1[4] = 0;                                    // tile_dim1 = 0 (unused)
      g1[5] = 1024;                                 // tensor_dim0_stride = 1024
      g1[6] = 0; g1[7] = 0;
      i32x4 z4 = {0, 0, 0, 0};
      i32x8 z8 = {0, 0, 0, 0, 0, 0, 0, 0};
      __builtin_amdgcn_tensor_load_to_lds(g0, g1, z4, z4, z8, 0);
    }
    __builtin_amdgcn_s_wait_tensorcnt(0);
  }
  __syncthreads();
#else
  for (int i = tid; i < 4 * CCH * CCH; i += 256) {
    int w = i >> 10; int rem = i & 1023;
    const float* W = (w == 0) ? Wq : (w == 1) ? Wk : (w == 2) ? Wv : Ws;
    Wlf[w][rem >> 5][rem & 31] = W[rem];
  }
  __syncthreads();
#endif

  int lane = tid & 31;
  int wave = tid >> 5;
  int half = lane >> 4;
  int rloc = lane & 15;

  // Persistent B fragments: B[32x16], VGPR j holds K=2j+8*half(+8 if j>=4),+1; N=lane&15
  v16bf bf[4][2];
#pragma unroll
  for (int w = 0; w < 4; ++w)
#pragma unroll
    for (int cb = 0; cb < 2; ++cb) {
      int colN = cb * 16 + rloc;
#pragma unroll
      for (int j = 0; j < 8; ++j) {
        int kin = 2 * j + 8 * half + (j >= 4 ? 8 : 0);
        bf[w][cb][2 * j]     = (__bf16)Wlf[w][kin][colN];
        bf[w][cb][2 * j + 1] = (__bf16)Wlf[w][kin + 1][colN];
      }
    }

  int rows = n * 3;
  int tiles = rows >> 4;
  int stride = gridDim.x * 8;
  float* outs[4] = {oq, ok, ov, osk};

  for (int tile = blockIdx.x * 8 + wave; tile < tiles; tile += stride) {
    int rowbase = tile * 16;
    const float* xr = x + (size_t)(rowbase + rloc) * CCH;
    // A fragment: lane row = rloc; half selects K chunks [8h,8h+8) and [16+8h,24+8h)
    float4 cA0 = *(const float4*)(xr + 8 * half);
    float4 cA1 = *(const float4*)(xr + 8 * half + 4);
    float4 cB0 = *(const float4*)(xr + 16 + 8 * half);
    float4 cB1 = *(const float4*)(xr + 16 + 8 * half + 4);
    v16bf a;
    a[0] = (__bf16)cA0.x; a[1] = (__bf16)cA0.y; a[2] = (__bf16)cA0.z; a[3] = (__bf16)cA0.w;
    a[4] = (__bf16)cA1.x; a[5] = (__bf16)cA1.y; a[6] = (__bf16)cA1.z; a[7] = (__bf16)cA1.w;
    a[8]  = (__bf16)cB0.x; a[9]  = (__bf16)cB0.y; a[10] = (__bf16)cB0.z; a[11] = (__bf16)cB0.w;
    a[12] = (__bf16)cB1.x; a[13] = (__bf16)cB1.y; a[14] = (__bf16)cB1.z; a[15] = (__bf16)cB1.w;

#pragma unroll
    for (int w = 0; w < 4; ++w) {
#pragma unroll
      for (int cb = 0; cb < 2; ++cb) {
        v8f c = {};
        c = __builtin_amdgcn_wmma_f32_16x16x32_bf16(false, a, false, bf[w][cb],
                                                    (short)0, c, false, false);
        // D: VGPR rr, lane L -> row rr + 8*(L>>4), col (L&15); coalesced stores
        float* op = outs[w] + (size_t)rowbase * CCH + cb * 16 + rloc;
#pragma unroll
        for (int rr = 0; rr < 8; ++rr)
          op[(size_t)(rr + 8 * half) * CCH] = c[rr];
      }
    }
  }
}

// ---------------------------------------------------------------------------
// First layer (Cin=1): out[(n*3+v)*32 + o] = v0[n*3+v] * W[o]   (coalesced in o)
// ---------------------------------------------------------------------------
__global__ void k_linear_first(const float* __restrict__ v0,
                               const float* __restrict__ Wq, const float* __restrict__ Wk,
                               const float* __restrict__ Wv, const float* __restrict__ Ws,
                               float* __restrict__ oq, float* __restrict__ ok,
                               float* __restrict__ ov, float* __restrict__ osk, int n) {
  int i = blockIdx.x * blockDim.x + threadIdx.x;
  if (i >= n * CCH) return;
  int node = i >> 5, o = i & 31;
  float x0 = v0[node * 3 + 0], x1 = v0[node * 3 + 1], x2 = v0[node * 3 + 2];
  size_t base = (size_t)node * (CCH * 3) + o;
  float wq = Wq[o], wk = Wk[o], wv = Wv[o], ws = Ws[o];
  oq[base] = x0 * wq; oq[base + CCH] = x1 * wq; oq[base + 2 * CCH] = x2 * wq;
  ok[base] = x0 * wk; ok[base + CCH] = x1 * wk; ok[base + 2 * CCH] = x2 * wk;
  ov[base] = x0 * wv; ov[base + CCH] = x1 * wv; ov[base + 2 * CCH] = x2 * wv;
  osk[base] = x0 * ws; osk[base + CCH] = x1 * ws; osk[base + 2 * CCH] = x2 * ws;
}

// Last layer (Cout=1): out[n*3+v] = sum_c x[(n*3+v)*32+c] * W[c]
__global__ void k_linear_last(const float* __restrict__ x,
                              const float* __restrict__ Wq, const float* __restrict__ Wk,
                              const float* __restrict__ Wv, const float* __restrict__ Ws,
                              float* __restrict__ oq, float* __restrict__ ok,
                              float* __restrict__ ov, float* __restrict__ osk, int n) {
  int node = blockIdx.x * blockDim.x + threadIdx.x;
  if (node >= n) return;
  float aq[3] = {0, 0, 0}, ak[3] = {0, 0, 0}, av[3] = {0, 0, 0}, as[3] = {0, 0, 0};
  const float* xr = x + (size_t)node * (CCH * 3);
#pragma unroll 4
  for (int c = 0; c < CCH; ++c) {
    float wq = Wq[c], wk = Wk[c], wv = Wv[c], ws = Ws[c];
#pragma unroll
    for (int vv = 0; vv < 3; ++vv) {
      float xv = xr[vv * CCH + c];
      aq[vv] += xv * wq; ak[vv] += xv * wk; av[vv] += xv * wv; as[vv] += xv * ws;
    }
  }
#pragma unroll
  for (int vv = 0; vv < 3; ++vv) {
    oq[node * 3 + vv] = aq[vv]; ok[node * 3 + vv] = ak[vv];
    ov[node * 3 + vv] = av[vv]; osk[node * 3 + vv] = as[vv];
  }
}

// ---------------------------------------------------------------------------
// Attention edge pipeline (generic over channel count Cc)
// ---------------------------------------------------------------------------
__global__ void k_logits(const float* __restrict__ q, const float* __restrict__ k,
                         const int* __restrict__ src, const int* __restrict__ dst,
                         const float* __restrict__ r,
                         const float* __restrict__ w1, const float* __restrict__ w2,
                         float* __restrict__ logits, int E, int Cc, float invs) {
  int e = blockIdx.x * blockDim.x + threadIdx.x;
  if (e >= E) return;
  int s = src[e], d = dst[e];
  const float* qd = q + (size_t)d * Cc * 3;
  const float* ks = k + (size_t)s * Cc * 3;
  float acc = 0.f;
  for (int i = 0; i < Cc * 3; ++i) acc += qd[i] * ks[i];  // layout-invariant dot
  acc *= invs;
  float rr = r[e], bias = 0.f;
#pragma unroll
  for (int j = 0; j < 16; ++j) {
    float h = rr * w1[j];
    bias += (h > 0.f ? h : 0.f) * w2[j];
  }
  logits[e] = acc + bias;
}

__global__ void k_init_mz(float* __restrict__ m, float* __restrict__ z, int n) {
  int i = blockIdx.x * blockDim.x + threadIdx.x;
  if (i >= n) return;
  m[i] = -3.4e38f;
  z[i] = 0.f;
}

__device__ void atomicMaxF(float* addr, float val) {
  int* ai = (int*)addr;
  int old = __float_as_int(*addr);
  while (__int_as_float(old) < val) {
    int assumed = old;
    old = atomicCAS(ai, assumed, __float_as_int(val));
    if (old == assumed) break;
  }
}

__global__ void k_segmax(const float* __restrict__ logits, const int* __restrict__ dst,
                         float* __restrict__ m, int E) {
  int e = blockIdx.x * blockDim.x + threadIdx.x;
  if (e >= E) return;
  atomicMaxF(&m[dst[e]], logits[e]);
}

__global__ void k_expz(float* __restrict__ logits, const int* __restrict__ dst,
                       const float* __restrict__ m, float* __restrict__ z, int E) {
  int e = blockIdx.x * blockDim.x + threadIdx.x;
  if (e >= E) return;
  int d = dst[e];
  float ex = __expf(logits[e] - m[d]);
  logits[e] = ex;
  atomicAdd(&z[d], ex);
}

__global__ void k_zero4(float4* __restrict__ p, int n4) {
  int i = blockIdx.x * blockDim.x + threadIdx.x;
  if (i < n4) p[i] = make_float4(0.f, 0.f, 0.f, 0.f);
}

__global__ void k_copy4(const float4* __restrict__ in, float4* __restrict__ out, int n4) {
  int i = blockIdx.x * blockDim.x + threadIdx.x;
  if (i < n4) out[i] = in[i];
}

__global__ void k_scatter_msg(const float* __restrict__ ex, const float* __restrict__ z,
                              const float* __restrict__ vf, const float* __restrict__ rel,
                              const int* __restrict__ src, const int* __restrict__ dst,
                              float* __restrict__ agg, int E, int Cc) {
  int i = blockIdx.x * blockDim.x + threadIdx.x;
  if (i >= E * Cc) return;
  int e = i / Cc, c = i - e * Cc;
  int s = src[e], d = dst[e];
  float alpha = ex[e] / (z[d] + EPSc);
  const float* vr = vf + (size_t)s * Cc * 3 + c;
  float* ar = agg + (size_t)d * Cc * 3 + c;
  const float* re = rel + (size_t)e * 3;
#pragma unroll
  for (int vv = 0; vv < 3; ++vv)
    atomicAdd(&ar[vv * Cc], (vr[vv * Cc] + re[vv]) * alpha);
}

__global__ void k_skip_normbias(const float* __restrict__ agg, const float* __restrict__ sk,
                                const float* __restrict__ scale, const float* __restrict__ bias,
                                float* __restrict__ out, int n, int Cc) {
  int i = blockIdx.x * blockDim.x + threadIdx.x;
  if (i >= n * Cc) return;
  int node = i / Cc, c = i - node * Cc;
  size_t base = (size_t)node * Cc * 3 + c;
  float t0 = agg[base] + sk[base];
  float t1 = agg[base + Cc] + sk[base + Cc];
  float t2 = agg[base + 2 * Cc] + sk[base + 2 * Cc];
  float nrm = sqrtf(t0 * t0 + t1 * t1 + t2 * t2 + 1e-12f);
  float val = scale[c] * nrm + bias[c];
  val = val > 0.f ? val : 0.f;
  float sfac = val / nrm;
  out[base] = t0 * sfac; out[base + Cc] = t1 * sfac; out[base + 2 * Cc] = t2 * sfac;
}

// ---------------------------------------------------------------------------
// Pooling / upsampling
// ---------------------------------------------------------------------------
__global__ void k_segacc(const float* __restrict__ x, const int* __restrict__ seg,
                         float* __restrict__ s, float* __restrict__ cnt, int n_src) {
  int i = blockIdx.x * blockDim.x + threadIdx.x;
  if (i >= n_src * CCH) return;
  int node = i >> 5, c = i & 31;
  int sg = seg[node];
  if (c == 0) atomicAdd(&cnt[sg], 1.f);
  const float* xr = x + (size_t)node * (CCH * 3) + c;
  float* sr = s + (size_t)sg * (CCH * 3) + c;
#pragma unroll
  for (int vv = 0; vv < 3; ++vv) atomicAdd(&sr[vv * CCH], xr[vv * CCH]);
}

__global__ void k_segdiv(float* __restrict__ s, const float* __restrict__ cnt, int n_dst) {
  int i = blockIdx.x * blockDim.x + threadIdx.x;
  if (i >= n_dst * CCH) return;
  int node = i >> 5, c = i & 31;
  float inv = 1.f / (cnt[node] + EPSc);
  size_t base = (size_t)node * (CCH * 3) + c;
  s[base] *= inv; s[base + CCH] *= inv; s[base + 2 * CCH] *= inv;
}

__global__ void k_add_at(const float* __restrict__ xsrc, const int* __restrict__ idx,
                         float* __restrict__ out, int n_src) {
  int i = blockIdx.x * blockDim.x + threadIdx.x;
  if (i >= n_src * CCH) return;
  int node = i >> 5, c = i & 31;
  int d = idx[node];
  const float* xr = xsrc + (size_t)node * (CCH * 3) + c;
  float* orow = out + (size_t)d * (CCH * 3) + c;
#pragma unroll
  for (int vv = 0; vv < 3; ++vv) orow[vv * CCH] += xr[vv * CCH];
}

// MLP head: y[n,o] = relu(sum_f x[n*3+f] * Wmlp[f,o])  (Cc=1 rows layout == n*3+f)
__global__ void k_mlp(const float* __restrict__ x, const float* __restrict__ W,
                      float* __restrict__ y, int n) {
  int i = blockIdx.x * blockDim.x + threadIdx.x;
  if (i >= n * OUTc) return;
  int node = i / OUTc, o = i - node * OUTc;
  float acc = x[node * 3 + 0] * W[0 * OUTc + o] +
              x[node * 3 + 1] * W[1 * OUTc + o] +
              x[node * 3 + 2] * W[2 * OUTc + o];
  y[i] = acc > 0.f ? acc : 0.f;
}

// ---------------------------------------------------------------------------
extern "C" void kernel_launch(void* const* d_in, const int* in_sizes, int n_in,
                              void* d_out, int out_size, void* d_ws, size_t ws_size,
                              hipStream_t stream) {
  const float* pos0   = (const float*)d_in[0];
  const float* v0     = (const float*)d_in[1];
  const float* Wqf    = (const float*)d_in[2];
  const float* Wkf    = (const float*)d_in[3];
  const float* Wvf    = (const float*)d_in[4];
  const float* Wsf    = (const float*)d_in[5];
  const float* Wqm    = (const float*)d_in[6];
  const float* Wkm    = (const float*)d_in[7];
  const float* Wvm    = (const float*)d_in[8];
  const float* Wsm    = (const float*)d_in[9];
  const float* Wql    = (const float*)d_in[10];
  const float* Wkl    = (const float*)d_in[11];
  const float* Wvl    = (const float*)d_in[12];
  const float* Wsl    = (const float*)d_in[13];
  const float* wr1    = (const float*)d_in[14];
  const float* wr2    = (const float*)d_in[15];
  const float* sc_mid = (const float*)d_in[16];
  const float* bi_mid = (const float*)d_in[17];
  const float* sc_lst = (const float*)d_in[18];
  const float* bi_lst = (const float*)d_in[19];
  const float* Wmlp   = (const float*)d_in[20];
  const int* src0 = (const int*)d_in[21];
  const int* dst0 = (const int*)d_in[22];
  const int* src1 = (const int*)d_in[23];
  const int* dst1 = (const int*)d_in[24];
  const int* src2 = (const int*)d_in[25];
  const int* dst2 = (const int*)d_in[26];
  const int* fp1  = (const int*)d_in[27];
  const int* fp2  = (const int*)d_in[28];
  const int* cl1  = (const int*)d_in[29];
  const int* cl2  = (const int*)d_in[30];
  float* yout = (float*)d_out;

  float* ws = (float*)d_ws;
  size_t off = 0;
  auto alloc = [&](size_t nf) { float* p = ws + off; off += nf; return p; };
  const size_t NB0 = (size_t)N0c * CCH * 3;
  float* qb   = alloc(NB0);
  float* kb   = alloc(NB0);
  float* vb   = alloc(NB0);
  float* skb  = alloc(NB0);
  float* agg  = alloc(NB0);
  float* xA   = alloc(NB0);
  float* xB   = alloc(NB0);
  float* x0s  = alloc(NB0);
  float* x1s  = alloc((size_t)N1c * CCH * 3);
  float* rel0 = alloc((size_t)E0c * 3);
  float* r0   = alloc(E0c);
  float* rel1 = alloc((size_t)E1c * 3);
  float* r1   = alloc(E1c);
  float* rel2 = alloc((size_t)E2c * 3);
  float* r2   = alloc(E2c);
  float* pos1 = alloc((size_t)N1c * 3);
  float* pos2 = alloc((size_t)N2c * 3);
  float* lg   = alloc(E0c);
  float* mb   = alloc(N0c);
  float* zb   = alloc(N0c);
  float* cnt  = alloc(N1c);

  const int T = 256;
  const float inv_mid = 1.f / sqrtf(3.f * 32.f);
  const float inv_lst = 1.f / sqrtf(3.f * 1.f);

  k_edge_geo<<<cdiv(E0c, T), T, 0, stream>>>(pos0, src0, dst0, rel0, r0, E0c);
  k_gather_pos<<<cdiv(N1c, T), T, 0, stream>>>(pos0, fp1, pos1, N1c);
  k_edge_geo<<<cdiv(E1c, T), T, 0, stream>>>(pos1, src1, dst1, rel1, r1, E1c);
  k_gather_pos<<<cdiv(N2c, T), T, 0, stream>>>(pos1, fp2, pos2, N2c);
  k_edge_geo<<<cdiv(E2c, T), T, 0, stream>>>(pos2, src2, dst2, rel2, r2, E2c);

  auto attn_tail = [&](int li, int Cc, int n, int Eg,
                       const int* src, const int* dst,
                       const float* rel, const float* r, float invs,
                       const float* scale, const float* bias, float* xout) {
    k_logits<<<cdiv(Eg, T), T, 0, stream>>>(qb, kb, src, dst, r,
                                            wr1 + li * 16, wr2 + li * 16, lg, Eg, Cc, invs);
    k_init_mz<<<cdiv(n, T), T, 0, stream>>>(mb, zb, n);
    k_segmax<<<cdiv(Eg, T), T, 0, stream>>>(lg, dst, mb, Eg);
    k_expz<<<cdiv(Eg, T), T, 0, stream>>>(lg, dst, mb, zb, Eg);
    k_zero4<<<cdiv((long long)n * Cc * 3 / 4, T), T, 0, stream>>>((float4*)agg, n * Cc * 3 / 4);
    k_scatter_msg<<<cdiv((long long)Eg * Cc, T), T, 0, stream>>>(lg, zb, vb, rel, src, dst, agg, Eg, Cc);
    k_skip_normbias<<<cdiv((long long)n * Cc, T), T, 0, stream>>>(agg, skb, scale, bias, xout, n, Cc);
  };

  auto mid_layer = [&](int li, int wj, int n, int Eg,
                       const int* src, const int* dst,
                       const float* rel, const float* r,
                       const float* xin, float* xout) {
    int tiles = (n * 3) / 16;
    k_linear4_wmma<<<cdiv(tiles, 8 * TPW), 256, 0, stream>>>(
        xin, Wqm + (size_t)wj * 1024, Wkm + (size_t)wj * 1024,
        Wvm + (size_t)wj * 1024, Wsm + (size_t)wj * 1024, qb, kb, vb, skb, n);
    attn_tail(li, CCH, n, Eg, src, dst, rel, r, inv_mid,
              sc_mid + li * CCH, bi_mid + li * CCH, xout);
  };

  auto segmean = [&](const float* xin, const int* seg, int n_src, int n_dst, float* xout) {
    k_zero4<<<cdiv((long long)n_dst * CCH * 3 / 4, T), T, 0, stream>>>((float4*)xout, n_dst * CCH * 3 / 4);
    k_zero4<<<cdiv(n_dst / 4, T), T, 0, stream>>>((float4*)cnt, n_dst / 4);
    k_segacc<<<cdiv((long long)n_src * CCH, T), T, 0, stream>>>(xin, seg, xout, cnt, n_src);
    k_segdiv<<<cdiv((long long)n_dst * CCH, T), T, 0, stream>>>(xout, cnt, n_dst);
  };

  auto upsample = [&](const float* xsrc, const int* idx, int n_src,
                      const float* xskip, int n_dst, float* xout) {
    k_copy4<<<cdiv((long long)n_dst * CCH * 3 / 4, T), T, 0, stream>>>(
        (const float4*)xskip, (float4*)xout, n_dst * CCH * 3 / 4);
    k_add_at<<<cdiv((long long)n_src * CCH, T), T, 0, stream>>>(xsrc, idx, xout, n_src);
  };

  // layer 0: first (Cin=1 -> 32), level 0
  k_linear_first<<<cdiv((long long)N0c * CCH, T), T, 0, stream>>>(
      v0, Wqf, Wkf, Wvf, Wsf, qb, kb, vb, skb, N0c);
  attn_tail(0, CCH, N0c, E0c, src0, dst0, rel0, r0, inv_mid, sc_mid, bi_mid, xA);

  // layer 1 -> x0 (residual save)
  mid_layer(1, 0, N0c, E0c, src0, dst0, rel0, r0, xA, x0s);
  // pool to level 1
  segmean(x0s, cl1, N0c, N1c, xA);
  // layers 2,3 (level 1); layer 3 output is x1 residual
  mid_layer(2, 1, N1c, E1c, src1, dst1, rel1, r1, xA, xB);
  mid_layer(3, 2, N1c, E1c, src1, dst1, rel1, r1, xB, x1s);
  // pool to level 2
  segmean(x1s, cl2, N1c, N2c, xA);
  // bottleneck layers 4,5 (level 2)
  mid_layer(4, 3, N2c, E2c, src2, dst2, rel2, r2, xA, xB);
  mid_layer(5, 4, N2c, E2c, src2, dst2, rel2, r2, xB, xA);
  // upsample to level 1 + skip, layers 6,7
  upsample(xA, fp2, N2c, x1s, N1c, xB);
  mid_layer(6, 5, N1c, E1c, src1, dst1, rel1, r1, xB, xA);
  mid_layer(7, 6, N1c, E1c, src1, dst1, rel1, r1, xA, xB);
  // upsample to level 0 + skip, layer 8
  upsample(xB, fp1, N1c, x0s, N0c, xA);
  mid_layer(8, 7, N0c, E0c, src0, dst0, rel0, r0, xA, xB);
  // layer 9: last (32 -> 1)
  k_linear_last<<<cdiv(N0c, T), T, 0, stream>>>(xB, Wql, Wkl, Wvl, Wsl, qb, kb, vb, skb, N0c);
  attn_tail(9, 1, N0c, E0c, src0, dst0, rel0, r0, inv_lst, sc_lst, bi_lst, xA);

  // MLP head
  k_mlp<<<cdiv((long long)N0c * OUTc, T), T, 0, stream>>>(xA, Wmlp, yout, N0c);
}
